// QuantLinear_19370302505132
// MI455X (gfx1250) — 8-bit matrix operand order repaired and checked by reading, compiled, not run on hardware
//
#include <hip/hip_runtime.h>

typedef __attribute__((ext_vector_type(8))) int v8i;

__global__ __launch_bounds__(256, 2)
void quant_rows_kernel(const float* __restrict__ src, signed char* __restrict__ dst,
                       float* __restrict__ scales, int K, float qmax)
{
    __shared__ float wmax[8];
    __shared__ float srowscale;

    const int row = blockIdx.x;
    const float* rp = src + (size_t)row * K;

    float amax = 0.0f;
    for (int i = threadIdx.x * 4; i < K; i += blockDim.x * 4) {
        float4 v = *(const float4*)(rp + i);
        amax = fmaxf(amax, fmaxf(fmaxf(fabsf(v.x), fabsf(v.y)),
                                 fmaxf(fabsf(v.z), fabsf(v.w))));
    }
    #pragma unroll
    for (int off = 16; off > 0; off >>= 1)
        amax = fmaxf(amax, __shfl_xor(amax, off, 32));

    const int lane = threadIdx.x & 31;
    const int wv   = threadIdx.x >> 5;
    if (lane == 0) wmax[wv] = amax;
    __syncthreads();
    if (threadIdx.x == 0) {
        float m = wmax[0];
        #pragma unroll
        for (int i = 1; i < 8; ++i) m = fmaxf(m, wmax[i]);
        float sc = fmaxf(m / qmax, 1e-8f);
        srowscale = sc;
        scales[row] = sc;
    }
    __syncthreads();

    const float sc = srowscale;
    const float lo = -qmax - 1.0f, hi = qmax;
    signed char* dp = dst + (size_t)row * K;
    for (int i = threadIdx.x * 4; i < K; i += blockDim.x * 4) {
        float4 v = *(const float4*)(rp + i);
        char4 q;
        q.x = (signed char)(int)fminf(fmaxf(rintf(v.x / sc), lo), hi);
        q.y = (signed char)(int)fminf(fmaxf(rintf(v.y / sc), lo), hi);
        q.z = (signed char)(int)fminf(fmaxf(rintf(v.z / sc), lo), hi);
        q.w = (signed char)(int)fminf(fmaxf(rintf(v.w / sc), lo), hi);
        *(char4*)(dp + i) = q;
    }
}

__global__ __launch_bounds__(256, 1)
void wmma_gemm_iu8_kernel(const signed char* __restrict__ qx,
                          const signed char* __restrict__ qw,
                          const float* __restrict__ sx,
                          const float* __restrict__ sw,
                          const float* __restrict__ bias,
                          float* __restrict__ out,
                          int M, int N, int K)
{
    const int lane = threadIdx.x & 31;
    const int wave = threadIdx.x >> 5;
    const int half = lane >> 4;
    const int l15  = lane & 15;

    const int mWave = blockIdx.y * 128 + (wave & 1) * 64;
    const int nWave = blockIdx.x * 128 + (wave >> 1) * 32;

    const v8i vzero = {0, 0, 0, 0, 0, 0, 0, 0};
    v8i acc[4][2];
    #pragma unroll
    for (int i = 0; i < 4; ++i)
        #pragma unroll
        for (int j = 0; j < 2; ++j)
            acc[i][j] = vzero;

    for (int k0 = 0; k0 < K; k0 += 64) {
        v8i a[4], b[2];

        #pragma unroll
        for (int i = 0; i < 4; ++i) {
            const signed char* ap =
                qx + (size_t)(mWave + i * 16 + l15) * K + k0 + half * 16;
            int2* av = (int2*)&a[i];
            av[0] = *(const int2*)(ap + 0);
            av[1] = *(const int2*)(ap + 8);
            av[2] = *(const int2*)(ap + 32);
            av[3] = *(const int2*)(ap + 40);
        }

        #pragma unroll
        for (int j = 0; j < 2; ++j) {
            const signed char* bp =
                qw + (size_t)(nWave + j * 16 + l15) * K + k0 + half * 16;
            int4* bv = (int4*)&b[j];
            bv[0] = *(const int4*)(bp + 0);
            bv[1] = *(const int4*)(bp + 32);
        }

        #pragma unroll
        for (int i = 0; i < 4; ++i)
            #pragma unroll
            for (int j = 0; j < 2; ++j)
                acc[i][j] = __builtin_amdgcn_wmma_i32_16x16x64_iu8(
                     true, a[i],  true, b[j],
                    acc[i][j],  false,  false);
    }

    #pragma unroll
    for (int i = 0; i < 4; ++i) {
        const int rowBase = mWave + i * 16 + half * 8;
        float4 s0 = *(const float4*)(sx + rowBase);
        float4 s1 = *(const float4*)(sx + rowBase + 4);
        float sxv[8] = {s0.x, s0.y, s0.z, s0.w, s1.x, s1.y, s1.z, s1.w};
        #pragma unroll
        for (int j = 0; j < 2; ++j) {
            const int col = nWave + j * 16 + l15;
            const float swv = sw[col];
            const float bv  = bias[col];
            #pragma unroll
            for (int r = 0; r < 8; ++r) {
                const int row = rowBase + r;
                out[(size_t)row * N + col] =
                    (float)acc[i][j][r] * sxv[r] * swv + bv;
            }
        }
    }
}

extern "C" void kernel_launch(void* const* d_in, const int* in_sizes, int n_in,
                              void* d_out, int out_size, void* d_ws, size_t ws_size,
                              hipStream_t stream)
{
    const float* x    = (const float*)d_in[0];
    const float* w    = (const float*)d_in[1];
    const float* bias = (const float*)d_in[2];
    float* out = (float*)d_out;

    const int D_OUT = in_sizes[2];
    const int D_IN  = in_sizes[1] / D_OUT;
    const int M     = in_sizes[0] / D_IN;

    signed char* qx = (signed char*)d_ws;
    signed char* qw = qx + (size_t)M * D_IN;
    float* sx = (float*)(qw + (size_t)D_OUT * D_IN);
    float* sw = sx + M;

    quant_rows_kernel<<<M, 256, 0, stream>>>(x, qx, sx, D_IN, 127.0f);
    quant_rows_kernel<<<D_OUT, 256, 0, stream>>>(w, qw, sw, D_IN, 7.0f);

    dim3 grid(D_OUT / 128, M / 128);
    wmma_gemm_iu8_kernel<<<grid, 256, 0, stream>>>(qx, qw, sx, sw, bias, out,
                                                   M, D_OUT, D_IN);
}
